// DSDModules_57681410785615
// MI455X (gfx1250) — matrix operand order repaired and checked by reading, not compiled, not run on hardware
//
#include <hip/hip_runtime.h>

typedef __attribute__((ext_vector_type(16))) _Float16 v16h;
typedef __attribute__((ext_vector_type(8)))  float    v8f;

// Problem constants
#define GG 8
#define KK 6
#define HH 64
#define WW 64
#define HW 4096
#define PADW 66
#define PADHW 4356   // 66*66

// ---------------------------------------------------------------------------
// Workspace layout (bytes, all multiples of 256)
// ---------------------------------------------------------------------------
#define OFF_INP  ((size_t)0)                     // 512*4356*2     = 4,460,544
#define OFF_HID1 ((size_t)4460544)               // 8*128*4356*2   = 8,921,088
#define OFF_HID2 ((size_t)13381632)              // 8*64*4356*2    = 4,460,544
#define OFF_HID3 ((size_t)17842176)              // 8*32*4356*2    = 2,230,272
#define OFF_OA   ((size_t)20072448)              // 8*18*4096*4    = 2,359,296
#define OFF_AP1  ((size_t)22431744)              // 8*4608*128*2   = 9,437,184
#define OFF_AP2  ((size_t)31868928)              // 8*1152*64*2    = 1,179,648
#define OFF_AP3  ((size_t)33048576)              // 8*576*32*2     =   294,912
#define OFF_AP4  ((size_t)33343488)              // 8*288*32*2     =   147,456
#define HID_U64  ((size_t)1951488)               // (8.92M+4.46M+2.23M)/8

// ---------------------------------------------------------------------------
// Pack concat(gar,cond) -> f16, zero-padded [512][66][66]
// ---------------------------------------------------------------------------
__global__ __launch_bounds__(256)
void pack_input_kernel(const float* __restrict__ gar, const float* __restrict__ cond,
                       _Float16* __restrict__ dst) {
  int idx = blockIdx.x * 256 + threadIdx.x;
  if (idx >= 512 * PADHW) return;
  int c = idx / PADHW;
  int p = idx - c * PADHW;
  int yy = p / PADW, xx = p - yy * PADW;
  float v = 0.0f;
  if (yy >= 1 && yy <= HH && xx >= 1 && xx <= WW) {
    int off = (yy - 1) * WW + (xx - 1);
    v = (c < 256) ? gar[c * HW + off] : cond[(c - 256) * HW + off];
  }
  dst[idx] = (_Float16)v;
}

__global__ __launch_bounds__(256)
void zero_u64_kernel(unsigned long long* __restrict__ p, size_t n) {
  size_t i = (size_t)blockIdx.x * 256 + threadIdx.x;
  if (i < n) p[i] = 0ull;
}

// ---------------------------------------------------------------------------
// Pre-pack weights [G][COUTR][CIN][3][3] f32 -> A-fragment order f16, k contiguous per lane half.
// K ordering: k = (r*3+s)*CIN + c.  Frag = 32 lanes x 16 halves (512 f16).
// Lane L (<16): M=mb*16+L, K = 0..15; lane L+16: same M, K = 16..31.
// ---------------------------------------------------------------------------
template<int CIN, int COUTP, int COUTR>
__global__ __launch_bounds__(256)
void prepack_kernel(const float* __restrict__ W, _Float16* __restrict__ Ap) {
  constexpr int KB = CIN * 9 / 32;
  constexpr int MB = COUTP / 16;
  const size_t total = (size_t)GG * KB * MB * 512;
  size_t idx = (size_t)blockIdx.x * 256 + threadIdx.x;
  if (idx >= total) return;
  int e    = (int)(idx & 511);
  int lane = e >> 4;
  int j    = e & 15;
  size_t frag = idx >> 9;
  int mb = (int)(frag % MB);
  int kb = (int)((frag / MB) % KB);
  int g  = (int)(frag / ((size_t)MB * KB));
  int base = (lane < 16) ? 0 : 16;
  int kin  = base + j;
  int k = kb * 32 + kin;
  int m = mb * 16 + (lane & 15);
  int rs = k / CIN, c = k - rs * CIN;
  int r = rs / 3, s = rs - r * 3;
  float v = 0.0f;
  if (m < COUTR)
    v = W[(((size_t)(g * COUTR + m) * CIN + c) * 3 + r) * 3 + s];
  Ap[idx] = (_Float16)v;
}

// ---------------------------------------------------------------------------
// Implicit-GEMM conv via v_wmma_f32_16x16x32_f16, double-buffered pipeline.
// One workgroup (8 waves) = (group g, 2 image rows): M=COUTP x N=128 x K=CIN*9.
// B tile staged in LDS pixel-major: [2 bufs][128 px][32 k-halves] (16 KB).
// Per K-step: prefetch next B tile + next A frag from global while WMMAs
// consume the current LDS buffer; single barrier per K-step.
// ---------------------------------------------------------------------------
template<int CIN, int COUTP, int COUTR, bool LRELU, bool OUTF32>
__global__ __launch_bounds__(256)
void conv_wmma_kernel(const _Float16* __restrict__ in, size_t gstride,
                      const _Float16* __restrict__ Ap,
                      const float* __restrict__ bias,
                      _Float16* __restrict__ out16,
                      float* __restrict__ out32) {
  constexpr int KB     = CIN * 9 / 32;
  constexpr int MB     = COUTP / 16;
  constexpr int WAVESM = MB;              // 8 / 4 / 2 / 2
  constexpr int WAVESN = 8 / WAVESM;      // 1 / 2 / 4 / 4
  constexpr int NSPAN  = 128 / WAVESN;    // per-wave pixel span
  constexpr int NFR    = NSPAN / 16;      // 8 / 4 / 2 / 2
  static_assert(WAVESM * WAVESN == 8 && NFR * WAVESN * 16 == 128, "tiling");

  __shared__ __align__(32) _Float16 lb[2][128 * 32];

  const int g   = blockIdx.x >> 5;        // 32 row-pair tiles
  const int y0  = (blockIdx.x & 31) * 2;
  const int tid = threadIdx.x;
  const int lane = tid & 31;
  const int w    = tid >> 5;
  const int mw = w % WAVESM;
  const int nw = w / WAVESM;
  const int n0 = nw * NSPAN;
  const int m0 = mw * 16;

  const _Float16* __restrict__ ing = in + (size_t)g * gstride;
  const _Float16* __restrict__ apg = Ap + (size_t)g * KB * MB * 512;

  v8f acc[NFR];
#pragma unroll
  for (int f = 0; f < NFR; ++f)
#pragma unroll
    for (int j = 0; j < 8; ++j) acc[f][j] = 0.0f;

  // cooperative-load coords: 32 channel rows x 8 pixel-groups of 8 (x 2 rows)
  const int kk = tid >> 3;
  const int xg = (tid & 7) * 8;

  auto gload = [&](int kb, _Float16* H0, _Float16* H1) {
    const int rs = (kb * 32) / CIN;
    const int c0 = (kb * 32) - rs * CIN;
    const int r = rs / 3, s = rs - r * 3;
    const _Float16* base = ing + (size_t)(c0 + kk) * PADHW + s + xg;
    __builtin_memcpy(H0, base + (size_t)(y0 + r) * PADW, 16);      // row y0
    __builtin_memcpy(H1, base + (size_t)(y0 + 1 + r) * PADW, 16);  // row y0+1
  };
  auto sstore = [&](int buf, const _Float16* H0, const _Float16* H1) {
#pragma unroll
    for (int i = 0; i < 8; ++i) {
      lb[buf][(xg + i) * 32 + kk]      = H0[i];
      lb[buf][(64 + xg + i) * 32 + kk] = H1[i];
    }
  };
  auto aload = [&](int kb) -> v16h {
    return *reinterpret_cast<const v16h*>(
        apg + ((size_t)kb * MB + mw) * 512 + lane * 16);
  };

  // pipeline prologue
  _Float16 h0[8], h1[8];
  gload(0, h0, h1);
  sstore(0, h0, h1);
  v16h a = aload(0);
  __syncthreads();

  for (int kb = 0; kb < KB; ++kb) {
    const int  cur  = kb & 1;
    const bool more = (kb + 1 < KB);
    _Float16 p0[8], p1[8];
    v16h an;
    if (more) {                 // issue global prefetches early
      gload(kb + 1, p0, p1);
      an = aload(kb + 1);
    }
#pragma unroll
    for (int f = 0; f < NFR; ++f) {
      const v16h b = *reinterpret_cast<const v16h*>(
          &lb[cur][(n0 + f * 16 + (lane & 15)) * 32 + (lane >> 4) * 16]);
      acc[f] = __builtin_amdgcn_wmma_f32_16x16x32_f16(
          false, a, false, b, (short)0, acc[f], false, false);
    }
    if (more) {
      sstore(1 - cur, p0, p1);  // other buffer: safe vs. concurrent readers
      a = an;
    }
    __syncthreads();
  }

  // Epilogue: C/D layout -> VGPR j = M row (j + 8*(lane>=16)), N col = lane&15
  const int ncol = lane & 15;
  const int mhi  = (lane >> 4) * 8;
#pragma unroll
  for (int j = 0; j < 8; ++j) {
    const int m = m0 + mhi + j;
    float bv = 0.0f;
    if (m < COUTR) bv = bias[g * COUTR + m];
#pragma unroll
    for (int f = 0; f < NFR; ++f) {
      float v = acc[f][j] + bv;
      if (LRELU) v = (v >= 0.0f) ? v : 0.1f * v;
      const int x   = n0 + f * 16 + ncol;      // 0..127
      const int row = y0 + (x >> 6);
      const int col = x & 63;
      if constexpr (OUTF32) {
        if (m < COUTR)
          out32[(size_t)(g * COUTR + m) * HW + row * WW + col] = v;
      } else {
        out16[(size_t)(g * COUTP + m) * PADHW + (size_t)(row + 1) * PADW + (col + 1)] =
            (_Float16)v;
      }
    }
  }
}

// ---------------------------------------------------------------------------
// Softmax-over-groups + offset grid + bilinear border sample + mask + sum.
// Thread = (pixel, 4-channel group). Consecutive threads = consecutive pixels.
// ---------------------------------------------------------------------------
__global__ __launch_bounds__(256)
void post_warp_kernel(const float* __restrict__ gar, const float* __restrict__ mask,
                      const float* __restrict__ oa, float* __restrict__ out) {
  const int t = blockIdx.x * 256 + threadIdx.x;
  if (t >= HW * 64) return;
  const int px = t & (HW - 1);
  const int cg = t >> 12;           // 0..63
  const int x = px & 63, y = px >> 6;
  const int c0 = cg * 4;

  // attention logits + stable softmax over groups (per ki)
  float lg[GG][KK];
#pragma unroll
  for (int g = 0; g < GG; ++g)
#pragma unroll
    for (int ki = 0; ki < KK; ++ki)
      lg[g][ki] = oa[(size_t)(g * 18 + 12 + ki) * HW + px];

  float mx[KK], invden[KK];
#pragma unroll
  for (int ki = 0; ki < KK; ++ki) {
    float m = lg[0][ki];
#pragma unroll
    for (int g = 1; g < GG; ++g) m = fmaxf(m, lg[g][ki]);
    float d = 0.0f;
#pragma unroll
    for (int g = 0; g < GG; ++g) d += __expf(lg[g][ki] - m);
    mx[ki] = m;
    invden[ki] = 1.0f / d;
  }

  float acc[4] = {0.f, 0.f, 0.f, 0.f};
  constexpr float SC = 64.0f / 63.0f;   // align_corners=False border map

#pragma unroll
  for (int g = 0; g < GG; ++g) {
    float tmp[4] = {0.f, 0.f, 0.f, 0.f};
#pragma unroll
    for (int ki = 0; ki < KK; ++ki) {
      const float a = __expf(lg[g][ki] - mx[ki]) * invden[ki];
      const float ox = oa[(size_t)(g * 18 + 2 * ki) * HW + px];
      const float oy = oa[(size_t)(g * 18 + 2 * ki + 1) * HW + px];
      float xs = fminf(fmaxf(((float)x + ox) * SC - 0.5f, 0.0f), 63.0f);
      float ys = fminf(fmaxf(((float)y + oy) * SC - 0.5f, 0.0f), 63.0f);
      float x0f = floorf(xs), y0f = floorf(ys);
      float wx = xs - x0f, wy = ys - y0f;
      int x0 = (int)x0f, y0 = (int)y0f;
      int x1 = min(x0 + 1, 63), y1 = min(y0 + 1, 63);
      float w00 = (1.f - wy) * (1.f - wx) * a;
      float w01 = (1.f - wy) * wx * a;
      float w10 = wy * (1.f - wx) * a;
      float w11 = wy * wx * a;
      const int i00 = y0 * 64 + x0, i01 = y0 * 64 + x1;
      const int i10 = y1 * 64 + x0, i11 = y1 * 64 + x1;
#pragma unroll
      for (int c = 0; c < 4; ++c) {
        const float* im = gar + (size_t)(c0 + c) * HW;
        tmp[c] += w00 * im[i00] + w01 * im[i01] + w10 * im[i10] + w11 * im[i11];
      }
    }
#pragma unroll
    for (int c = 0; c < 4; ++c) acc[c] += tmp[c] * mask[g * 256 + c0 + c];
  }

#pragma unroll
  for (int c = 0; c < 4; ++c) out[(size_t)(c0 + c) * HW + px] = acc[c];
}

// ---------------------------------------------------------------------------
// Launch
// ---------------------------------------------------------------------------
extern "C" void kernel_launch(void* const* d_in, const int* in_sizes, int n_in,
                              void* d_out, int out_size, void* d_ws, size_t ws_size,
                              hipStream_t stream) {
  const float* gar  = (const float*)d_in[0];
  const float* cond = (const float*)d_in[1];
  const float* mask = (const float*)d_in[2];
  const float* W1 = (const float*)d_in[3];
  const float* b1 = (const float*)d_in[4];
  const float* W2 = (const float*)d_in[5];
  const float* b2 = (const float*)d_in[6];
  const float* W3 = (const float*)d_in[7];
  const float* b3 = (const float*)d_in[8];
  const float* W4 = (const float*)d_in[9];
  const float* b4 = (const float*)d_in[10];

  char* ws = (char*)d_ws;
  _Float16* inp  = (_Float16*)(ws + OFF_INP);
  _Float16* hid1 = (_Float16*)(ws + OFF_HID1);
  _Float16* hid2 = (_Float16*)(ws + OFF_HID2);
  _Float16* hid3 = (_Float16*)(ws + OFF_HID3);
  float*    oa   = (float*)   (ws + OFF_OA);
  _Float16* Ap1  = (_Float16*)(ws + OFF_AP1);
  _Float16* Ap2  = (_Float16*)(ws + OFF_AP2);
  _Float16* Ap3  = (_Float16*)(ws + OFF_AP3);
  _Float16* Ap4  = (_Float16*)(ws + OFF_AP4);

  // 1) activation packing + hidden zero-init (pads must be 0 every call)
  pack_input_kernel<<<(512 * PADHW + 255) / 256, 256, 0, stream>>>(gar, cond, inp);
  zero_u64_kernel<<<(int)((HID_U64 + 255) / 256), 256, 0, stream>>>(
      (unsigned long long*)(ws + OFF_HID1), HID_U64);

  // 2) weight pre-pack into A-fragment layout (contiguous k per lane half)
  prepack_kernel<512, 128, 128><<<(8 * 144 * 8 * 512 + 255) / 256, 256, 0, stream>>>(W1, Ap1);
  prepack_kernel<128,  64,  64><<<(8 *  36 * 4 * 512 + 255) / 256, 256, 0, stream>>>(W2, Ap2);
  prepack_kernel< 64,  32,  32><<<(8 *  18 * 2 * 512 + 255) / 256, 256, 0, stream>>>(W3, Ap3);
  prepack_kernel< 32,  32,  18><<<(8 *   9 * 2 * 512 + 255) / 256, 256, 0, stream>>>(W4, Ap4);

  // 3) conv stack (grid = G * 32 row-pair tiles)
  conv_wmma_kernel<512, 128, 128, true,  false><<<GG * 32, 256, 0, stream>>>(
      inp, 0, Ap1, b1, hid1, nullptr);
  conv_wmma_kernel<128,  64,  64, true,  false><<<GG * 32, 256, 0, stream>>>(
      hid1, (size_t)128 * PADHW, Ap2, b2, hid2, nullptr);
  conv_wmma_kernel< 64,  32,  32, true,  false><<<GG * 32, 256, 0, stream>>>(
      hid2, (size_t)64 * PADHW, Ap3, b3, hid3, nullptr);
  conv_wmma_kernel< 32,  32,  18, false, true ><<<GG * 32, 256, 0, stream>>>(
      hid3, (size_t)32 * PADHW, Ap4, b4, nullptr, oa);

  // 4) softmax + warp + aggregate
  post_warp_kernel<<<(HW * 64) / 256, 256, 0, stream>>>(gar, mask, oa, (float*)d_out);
}
